// NNSwinTransformerBlock_39702677684799
// MI455X (gfx1250) — compile-verified, matrix operand order checked by reading, not run on hardware
//
#include <hip/hip_runtime.h>
#include <hip/hip_bf16.h>

// ---------------------------------------------------------------------------
// Swin 3D block for gfx1250 (MI455X). All matmuls on v_wmma_f32_16x16x32_f16.
// ---------------------------------------------------------------------------

typedef __attribute__((ext_vector_type(16))) _Float16 v16h;
typedef __attribute__((ext_vector_type(8)))  _Float16 v8h;
typedef __attribute__((ext_vector_type(8)))  float    v8f;

#define DIMC   192
#define HEADS  6
#define HDIM   32        // DIMC / HEADS
#define WS     8
#define SHIFT  4
#define RES    32
#define NB     2
#define NWIN   64        // (RES/WS)^3
#define BWIN   128       // NB * NWIN
#define NTOK   512       // WS^3
#define MTOT   65536     // NB * RES^3  == BWIN * NTOK
#define HID    768
#define QSCALE 0.17677669529663687f   // (DIMC/HEADS)^-0.5

// ------------------------------- fragments ---------------------------------
// 16-bit A / B^T fragment (ISA 7.12.2): lane holds row (lane&15), two
// contiguous 8-half runs at k0 + (lane>>4)*8 and k0 + 16 + (lane>>4)*8.
__device__ __forceinline__ v16h ldfrag(const _Float16* base, int ld, int row, int k0) {
  const int lane = threadIdx.x & 31;
  const _Float16* p = base + (size_t)(row + (lane & 15)) * ld + k0 + ((lane >> 4) << 3);
  v8h x0 = *(const v8h*)p;
  v8h x1 = *(const v8h*)(p + 16);
  v16h r;
#pragma unroll
  for (int i = 0; i < 8; ++i) { r[i] = x0[i]; r[i + 8] = x1[i]; }
  return r;
}

__device__ __forceinline__ v8f wmma16(v16h a, v16h b, v8f c) {
  return __builtin_amdgcn_wmma_f32_16x16x32_f16(false, a, false, b, (short)0, c, false, false);
}

// K-loop for one 16x16 output tile. A row-major [M,K]; W row-major [N,K] (=B^T).
template <int K>
__device__ __forceinline__ v8f gemm_acc(const _Float16* A, const _Float16* W,
                                        int row, int col, v8f acc) {
#pragma unroll
  for (int k0 = 0; k0 < K; k0 += 32) {
    v16h a = ldfrag(A, K, row, k0);
    v16h b = ldfrag(W, K, col, k0);
    acc = wmma16(a, b, acc);
  }
  return acc;
}

// ------------------------------- prep kernels ------------------------------
__global__ void k_f32_to_f16(const float* __restrict__ s, _Float16* __restrict__ d, int n) {
  int i = blockIdx.x * blockDim.x + threadIdx.x;
  if (i < n) d[i] = (_Float16)s[i];
}

// rpbT[h][n][m] = rpb_table[rel_idx[n,m]][h]
__global__ void k_build_rpbT(const float* __restrict__ table, const int* __restrict__ rel,
                             float* __restrict__ rpbT) {
  int i = blockIdx.x * blockDim.x + threadIdx.x;
  if (i >= NTOK * NTOK) return;
  int r = rel[i];
#pragma unroll
  for (int h = 0; h < HEADS; ++h)
    rpbT[(size_t)h * NTOK * NTOK + i] = table[r * HEADS + h];
}

// ------------------------- LN1 + shift + window gather ---------------------
// One wave per window-ordered token; lane owns 6 channels.
__global__ __launch_bounds__(256) void k_ln1_gather(
    const float* __restrict__ x, const float* __restrict__ w, const float* __restrict__ b,
    _Float16* __restrict__ xw) {
  int tw = blockIdx.x * (blockDim.x >> 5) + (threadIdx.x >> 5);
  int lane = threadIdx.x & 31;
  if (tw >= MTOT) return;
  int b_ = tw >> 9, tok = tw & 511;
  int bb = b_ >> 6, wi = b_ & 63;
  int ss = ((wi >> 4) & 3) * WS + (tok >> 6);
  int sh = ((wi >> 2) & 3) * WS + ((tok >> 3) & 7);
  int sw = (wi & 3) * WS + (tok & 7);
  int ns = (ss + SHIFT) & 31, nh = (sh + SHIFT) & 31, nw = (sw + SHIFT) & 31;
  size_t src = ((((size_t)bb * RES + ns) * RES + nh) * RES + nw) * DIMC;
  const float* xr = x + src;
  float v[6], s = 0.f;
#pragma unroll
  for (int i = 0; i < 6; ++i) { v[i] = xr[lane + 32 * i]; s += v[i]; }
#pragma unroll
  for (int o = 16; o; o >>= 1) s += __shfl_xor(s, o);
  float mu = s * (1.f / DIMC), var = 0.f;
#pragma unroll
  for (int i = 0; i < 6; ++i) { float d = v[i] - mu; var += d * d; }
#pragma unroll
  for (int o = 16; o; o >>= 1) var += __shfl_xor(var, o);
  float rstd = rsqrtf(var * (1.f / DIMC) + 1e-5f);
  _Float16* dst = xw + (size_t)tw * DIMC;
#pragma unroll
  for (int i = 0; i < 6; ++i) {
    int c = lane + 32 * i;
    dst[c] = (_Float16)((v[i] - mu) * rstd * w[c] + b[c]);
  }
}

// LN2 on natural-order h (f32) -> h16.
__global__ __launch_bounds__(256) void k_ln2(
    const float* __restrict__ h, const float* __restrict__ w, const float* __restrict__ b,
    _Float16* __restrict__ h16) {
  int t = blockIdx.x * (blockDim.x >> 5) + (threadIdx.x >> 5);
  int lane = threadIdx.x & 31;
  if (t >= MTOT) return;
  const float* xr = h + (size_t)t * DIMC;
  float v[6], s = 0.f;
#pragma unroll
  for (int i = 0; i < 6; ++i) { v[i] = xr[lane + 32 * i]; s += v[i]; }
#pragma unroll
  for (int o = 16; o; o >>= 1) s += __shfl_xor(s, o);
  float mu = s * (1.f / DIMC), var = 0.f;
#pragma unroll
  for (int i = 0; i < 6; ++i) { float d = v[i] - mu; var += d * d; }
#pragma unroll
  for (int o = 16; o; o >>= 1) var += __shfl_xor(var, o);
  float rstd = rsqrtf(var * (1.f / DIMC) + 1e-5f);
  _Float16* dst = h16 + (size_t)t * DIMC;
#pragma unroll
  for (int i = 0; i < 6; ++i) {
    int c = lane + 32 * i;
    dst[c] = (_Float16)((v[i] - mu) * rstd * w[c] + b[c]);
  }
}

// ------------------------------- QKV GEMM ----------------------------------
// out = xw @ qkv_w^T + qkv_b; scatter into q(*scale)[bh][tok][d], k[bh][tok][d],
// vT[bh][d][tok].
__global__ __launch_bounds__(256) void k_qkv_gemm(
    const _Float16* __restrict__ xw, const _Float16* __restrict__ w16,
    const float* __restrict__ bias, _Float16* __restrict__ qb,
    _Float16* __restrict__ kb, _Float16* __restrict__ vTb) {
  int wid = threadIdx.x >> 5, lane = threadIdx.x & 31;
  int hi = lane >> 4, lo = lane & 15;
  int row = (blockIdx.x * 4 + (wid >> 1)) * 16;
  int col = (blockIdx.y * 2 + (wid & 1)) * 16;
  float bv = bias[col + lo];
  v8f acc;
#pragma unroll
  for (int j = 0; j < 8; ++j) acc[j] = bv;
  acc = gemm_acc<DIMC>(xw, w16, row, col, acc);
  int o = col + lo;
  int sel = o / DIMC, head = (o % DIMC) >> 5, dd = o & 31;
#pragma unroll
  for (int j = 0; j < 8; ++j) {
    int m = row + 8 * hi + j;
    int b_ = m >> 9, tok = m & 511;
    size_t bh = (size_t)b_ * HEADS + head;
    float val = acc[j];
    if (sel == 0)      qb[(bh * NTOK + tok) * HDIM + dd] = (_Float16)(val * QSCALE);
    else if (sel == 1) kb[(bh * NTOK + tok) * HDIM + dd] = (_Float16)val;
    else               vTb[(bh * HDIM + dd) * NTOK + tok] = (_Float16)val;
  }
}

// ------------------------------ attention ----------------------------------
// One block per (window, head): 8 waves, wave handles 64 queries (4 x 16-row
// tiles), flash-style online softmax over 16 chunks of 32 keys.
__global__ __launch_bounds__(256) void k_attn(
    const _Float16* __restrict__ q, const _Float16* __restrict__ k,
    const _Float16* __restrict__ vT, const float* __restrict__ rpbT,
    const float* __restrict__ mask, _Float16* __restrict__ out) {
  __shared__ float    stile[8][16][33];      // padded vs bank conflicts
  __shared__ _Float16 ptile[8][16][32];
  __shared__ float    sscale[8][16];

  const int wid = threadIdx.x >> 5, lane = threadIdx.x & 31;
  const int hi = lane >> 4, lo = lane & 15;
  const int bh = blockIdx.x;                 // 0 .. BWIN*HEADS-1
  const int b_ = bh / HEADS, head = bh % HEADS;
  const int wdw = b_ & (NWIN - 1);
  const _Float16* Q = q  + (size_t)bh * NTOK * HDIM;
  const _Float16* K = k  + (size_t)bh * NTOK * HDIM;
  const _Float16* V = vT + (size_t)bh * HDIM * NTOK;
  const float* RP = rpbT + (size_t)head * NTOK * NTOK;
  const float* MK = mask + (size_t)wdw * NTOK * NTOK;

  for (int q0 = wid * 64; q0 < wid * 64 + 64; q0 += 16) {
    v8f o0{}, o1{};
    float rowM = -1e30f, rowL = 0.f;         // valid on lanes 0..15 (row = lane)
    v16h aq = ldfrag(Q, HDIM, q0, 0);        // Q tile constant per row-block
    for (int kc = 0; kc < NTOK; kc += 32) {
#pragma unroll
      for (int sub = 0; sub < 2; ++sub) {
        int kb = kc + sub * 16;
        int colk = kb + lo;
        v8f c;
#pragma unroll
        for (int j = 0; j < 8; ++j) {
          int r = q0 + 8 * hi + j;
          c[j] = RP[(size_t)r * NTOK + colk] + MK[(size_t)r * NTOK + colk];
        }
        v16h bf = ldfrag(K, HDIM, kb, 0);
        v8f s = wmma16(aq, bf, c);
#pragma unroll
        for (int j = 0; j < 8; ++j) stile[wid][8 * hi + j][sub * 16 + lo] = s[j];
      }
      asm volatile("s_wait_dscnt 0" ::: "memory");
      if (lane < 16) {                       // row pass (re-converges before wmma)
        float sv[32], tmax = -1e30f;
#pragma unroll
        for (int m = 0; m < 32; ++m) { sv[m] = stile[wid][lane][m]; tmax = fmaxf(tmax, sv[m]); }
        float newM = fmaxf(rowM, tmax);
        float sc = __expf(rowM - newM);
        float ps = 0.f;
#pragma unroll
        for (int m = 0; m < 32; ++m) {
          float p = __expf(sv[m] - newM);
          ps += p;
          ptile[wid][lane][m] = (_Float16)p;
        }
        rowL = rowL * sc + ps;
        rowM = newM;
        sscale[wid][lane] = sc;
      }
      asm volatile("s_wait_dscnt 0" ::: "memory");
#pragma unroll
      for (int j = 0; j < 8; ++j) {
        float f = sscale[wid][8 * hi + j];
        o0[j] *= f; o1[j] *= f;
      }
      v16h pf;
#pragma unroll
      for (int i = 0; i < 8; ++i) {
        pf[i]     = ptile[wid][lo][hi * 8 + i];
        pf[i + 8] = ptile[wid][lo][16 + hi * 8 + i];
      }
      v16h v0 = ldfrag(V, NTOK, 0,  kc);
      v16h v1 = ldfrag(V, NTOK, 16, kc);
      o0 = wmma16(pf, v0, o0);
      o1 = wmma16(pf, v1, o1);
    }
    if (lane < 16) sscale[wid][lane] = 1.f / rowL;
    asm volatile("s_wait_dscnt 0" ::: "memory");
    size_t base = ((size_t)b_ * NTOK) * DIMC + head * HDIM;
#pragma unroll
    for (int j = 0; j < 8; ++j) {
      float f = sscale[wid][8 * hi + j];
      size_t rb = base + (size_t)(q0 + 8 * hi + j) * DIMC;
      out[rb + lo]      = (_Float16)(o0[j] * f);
      out[rb + 16 + lo] = (_Float16)(o1[j] * f);
    }
  }
}

// --------------------- proj GEMM + reverse/roll + residual -----------------
__global__ __launch_bounds__(256) void k_proj_gemm(
    const _Float16* __restrict__ aout, const _Float16* __restrict__ w16,
    const float* __restrict__ bias, const float* __restrict__ x,
    float* __restrict__ h) {
  int wid = threadIdx.x >> 5, lane = threadIdx.x & 31;
  int hi = lane >> 4, lo = lane & 15;
  int row = (blockIdx.x * 4 + (wid >> 1)) * 16;
  int col = (blockIdx.y * 2 + (wid & 1)) * 16;
  float bv = bias[col + lo];
  v8f acc;
#pragma unroll
  for (int j = 0; j < 8; ++j) acc[j] = bv;
  acc = gemm_acc<DIMC>(aout, w16, row, col, acc);
  int cch = col + lo;
#pragma unroll
  for (int j = 0; j < 8; ++j) {
    int m = row + 8 * hi + j;
    int b_ = m >> 9, tok = m & 511;
    int bb = b_ >> 6, wi = b_ & 63;
    int ss = ((wi >> 4) & 3) * WS + (tok >> 6);
    int sh = ((wi >> 2) & 3) * WS + ((tok >> 3) & 7);
    int sw = (wi & 3) * WS + (tok & 7);
    int ns = (ss + SHIFT) & 31, nh = (sh + SHIFT) & 31, nw = (sw + SHIFT) & 31;
    size_t tn = (((size_t)bb * RES + ns) * RES + nh) * RES + nw;
    h[tn * DIMC + cch] = x[tn * DIMC + cch] + acc[j];
  }
}

// ----------------------------- fc1 GEMM + GELU -----------------------------
__global__ __launch_bounds__(256) void k_fc1_gemm(
    const _Float16* __restrict__ h16, const _Float16* __restrict__ w16,
    const float* __restrict__ bias, _Float16* __restrict__ g16) {
  int wid = threadIdx.x >> 5, lane = threadIdx.x & 31;
  int hi = lane >> 4, lo = lane & 15;
  int row = (blockIdx.x * 4 + (wid >> 1)) * 16;
  int col = (blockIdx.y * 2 + (wid & 1)) * 16;
  float bv = bias[col + lo];
  v8f acc;
#pragma unroll
  for (int j = 0; j < 8; ++j) acc[j] = bv;
  acc = gemm_acc<DIMC>(h16, w16, row, col, acc);
  int cch = col + lo;
#pragma unroll
  for (int j = 0; j < 8; ++j) {
    int m = row + 8 * hi + j;
    float v = acc[j];
    float g = 0.5f * v * (1.f + erff(v * 0.70710678118654752f));   // exact GELU
    g16[(size_t)m * HID + cch] = (_Float16)g;
  }
}

// ---------------------------- fc2 GEMM + residual --------------------------
__global__ __launch_bounds__(256) void k_fc2_gemm(
    const _Float16* __restrict__ g16, const _Float16* __restrict__ w16,
    const float* __restrict__ bias, const float* __restrict__ h,
    float* __restrict__ out) {
  int wid = threadIdx.x >> 5, lane = threadIdx.x & 31;
  int hi = lane >> 4, lo = lane & 15;
  int row = (blockIdx.x * 4 + (wid >> 1)) * 16;
  int col = (blockIdx.y * 2 + (wid & 1)) * 16;
  float bv = bias[col + lo];
  v8f acc;
#pragma unroll
  for (int j = 0; j < 8; ++j) acc[j] = bv;
  acc = gemm_acc<HID>(g16, w16, row, col, acc);
  int cch = col + lo;
#pragma unroll
  for (int j = 0; j < 8; ++j) {
    int m = row + 8 * hi + j;
    size_t idx = (size_t)m * DIMC + cch;
    out[idx] = h[idx] + acc[j];
  }
}

// -------------------------------- launcher ---------------------------------
extern "C" void kernel_launch(void* const* d_in, const int* in_sizes, int n_in,
                              void* d_out, int out_size, void* d_ws, size_t ws_size,
                              hipStream_t stream) {
  const float* x     = (const float*)d_in[0];
  const float* mask  = (const float*)d_in[1];
  const float* n1w   = (const float*)d_in[2];
  const float* n1b   = (const float*)d_in[3];
  const float* qkvw  = (const float*)d_in[4];
  const float* qkvb  = (const float*)d_in[5];
  const float* rpbt  = (const float*)d_in[6];
  const float* projw = (const float*)d_in[7];
  const float* projb = (const float*)d_in[8];
  const float* n2w   = (const float*)d_in[9];
  const float* n2b   = (const float*)d_in[10];
  const float* fc1w  = (const float*)d_in[11];
  const float* fc1b  = (const float*)d_in[12];
  const float* fc2w  = (const float*)d_in[13];
  const float* fc2b  = (const float*)d_in[14];
  const int*   relix = (const int*)d_in[15];
  float* out = (float*)d_out;
  char*  ws  = (char*)d_ws;

  // workspace layout (liveness-based reuse)
  const size_t SZ_A    = (size_t)MTOT * HID * 2;          // 100663296: xw -> attnout -> g16
  const size_t OFF_A   = 0;
  const size_t OFF_Q   = OFF_A + SZ_A;
  const size_t SZ_QKV1 = (size_t)BWIN * HEADS * NTOK * HDIM * 2;   // 25165824 each
  const size_t OFF_K   = OFF_Q + SZ_QKV1;
  const size_t OFF_VT  = OFF_K + SZ_QKV1;
  const size_t OFF_H   = OFF_VT + SZ_QKV1;                // h f32, 50331648
  const size_t OFF_RPT = OFF_H + (size_t)MTOT * DIMC * 4;
  const size_t OFF_W16 = OFF_RPT + (size_t)HEADS * NTOK * NTOK * 4;

  _Float16* xw    = (_Float16*)(ws + OFF_A);
  _Float16* aout  = (_Float16*)(ws + OFF_A);              // reuses xw
  _Float16* g16   = (_Float16*)(ws + OFF_A);              // reuses attnout
  _Float16* qbuf  = (_Float16*)(ws + OFF_Q);
  _Float16* kbuf  = (_Float16*)(ws + OFF_K);
  _Float16* vTbuf = (_Float16*)(ws + OFF_VT);
  _Float16* h16   = (_Float16*)(ws + OFF_Q);              // reuses q after attention
  float*    hbuf  = (float*)(ws + OFF_H);
  float*    rpbT  = (float*)(ws + OFF_RPT);
  _Float16* qkvw16 = (_Float16*)(ws + OFF_W16);
  _Float16* projw16 = qkvw16 + (size_t)3 * DIMC * DIMC;
  _Float16* fc1w16  = projw16 + (size_t)DIMC * DIMC;
  _Float16* fc2w16  = fc1w16 + (size_t)HID * DIMC;

  // 1) weight conversion + rpbT gather
  k_f32_to_f16<<<(3 * DIMC * DIMC + 255) / 256, 256, 0, stream>>>(qkvw, qkvw16, 3 * DIMC * DIMC);
  k_f32_to_f16<<<(DIMC * DIMC + 255) / 256, 256, 0, stream>>>(projw, projw16, DIMC * DIMC);
  k_f32_to_f16<<<(HID * DIMC + 255) / 256, 256, 0, stream>>>(fc1w, fc1w16, HID * DIMC);
  k_f32_to_f16<<<(HID * DIMC + 255) / 256, 256, 0, stream>>>(fc2w, fc2w16, HID * DIMC);
  k_build_rpbT<<<(NTOK * NTOK + 255) / 256, 256, 0, stream>>>(rpbt, relix, rpbT);

  // 2) LN1 + shift-roll + window partition (f16)
  k_ln1_gather<<<MTOT / 8, 256, 0, stream>>>(x, n1w, n1b, xw);

  // 3) QKV GEMM -> q (scaled) / k / vT
  {
    dim3 g(MTOT / 64, (3 * DIMC) / 32);
    k_qkv_gemm<<<g, 256, 0, stream>>>(xw, qkvw16, qkvb, qbuf, kbuf, vTbuf);
  }

  // 4) windowed attention (flash softmax, WMMA QK^T and PV)
  k_attn<<<BWIN * HEADS, 256, 0, stream>>>(qbuf, kbuf, vTbuf, rpbT, mask, aout);

  // 5) proj GEMM + window-reverse + roll + residual -> h (f32)
  {
    dim3 g(MTOT / 64, DIMC / 32);
    k_proj_gemm<<<g, 256, 0, stream>>>(aout, projw16, projb, x, hbuf);
  }

  // 6) LN2 -> h16
  k_ln2<<<MTOT / 8, 256, 0, stream>>>(hbuf, n2w, n2b, h16);

  // 7) fc1 + GELU
  {
    dim3 g(MTOT / 64, HID / 32);
    k_fc1_gemm<<<g, 256, 0, stream>>>(h16, fc1w16, fc1b, g16);
  }

  // 8) fc2 + residual -> out
  {
    dim3 g(MTOT / 64, DIMC / 32);
    k_fc2_gemm<<<g, 256, 0, stream>>>(g16, fc2w16, fc2b, hbuf, out);
  }
}
